// RegionSelection_57784490001007
// MI455X (gfx1250) — matrix operand order repaired and checked by reading, not compiled, not run on hardware
//
#include <hip/hip_runtime.h>
#include <hip/hip_bf16.h>

// ---------------------------------------------------------------------------
// RegionSelection on MI455X (gfx1250, wave32, WMMA)
//   attention MLP  -> bf16 V_WMMA_F32_16X16X32_BF16 GEMMs (fp32 accum)
//   layer1 staging -> GLOBAL_LOAD_ASYNC_TO_LDS_B32 double-buffered transpose
//   peak picking   -> LDS argmax + suppression
//   region crops   -> coalesced b128 HBM copy
// ---------------------------------------------------------------------------

typedef __attribute__((ext_vector_type(16))) __bf16 v16bf;
typedef __attribute__((ext_vector_type(8)))  __bf16 v8bf;
typedef __attribute__((ext_vector_type(8)))  float  v8f;
typedef __attribute__((ext_vector_type(4)))  float  v4f;

// 32x16 bf16 B-fragment from two contiguous 16-byte chunks (global bf16).
__device__ __forceinline__ v16bf load_frag_b(const __bf16* p0, const __bf16* p1) {
    v8bf lo = *(const v8bf*)p0;
    v8bf hi = *(const v8bf*)p1;
    v16bf r;
#pragma unroll
    for (int i = 0; i < 8; ++i) { r[i] = lo[i]; r[i + 8] = hi[i]; }
    return r;
}

// Per-lane async copy: global fp32 -> LDS (ASYNCcnt-tracked, no VGPR return).
__device__ __forceinline__ void async_cp_b32(unsigned lds_addr, const float* gptr) {
    unsigned long long ga = (unsigned long long)(const void*)gptr;
    asm volatile("global_load_async_to_lds_b32 %0, %1, off"
                 :: "v"(lds_addr), "v"(ga) : "memory");
}
__device__ __forceinline__ void wait_async0() {
    asm volatile("s_wait_asynccnt 0x0" ::: "memory");
}

// ----------------------------- fp32 -> bf16 --------------------------------
__global__ void cvt_bf16(const float* __restrict__ src, __bf16* __restrict__ dst, int n) {
    int i = blockIdx.x * blockDim.x + threadIdx.x;
    if (i < n) dst[i] = (__bf16)src[i];
}

// ----------------------------- layer 1 -------------------------------------
// h1[p,o] = relu( sum_c feat[b,c,p] * w1[o,c] + b1[o] ),  p=32768, o=512
// WG: 256 thr (8 waves) -> 32 pixels x 512 outputs. Waves: 2 pixel-rows x 4 o-groups.
// features are channel-major => async transpose-stage [32 px x 128 ch] fp32
// tiles into LDS (double buffered), convert to bf16 when building A fragments.
#define KC  128   // K chunk
#define XPF 132   // fp32 LDS pitch per pixel row (16B-aligned, staggered banks)

__global__ __launch_bounds__(256) void k_layer1(const float* __restrict__ feat,
                                                const __bf16* __restrict__ w1b,
                                                const float* __restrict__ b1,
                                                __bf16* __restrict__ h1) {
    __shared__ alignas(16) float XtF[2][32 * XPF];

    const int tid  = threadIdx.x;
    const int lane = tid & 31;
    const int wave = tid >> 5;
    const int wg   = blockIdx.x;          // 1024 WGs
    const int b    = wg >> 5;             // batch
    const int p0   = (wg & 31) * 32;      // pixel base inside batch (HW plane = 1024 px)
    const float* fb = feat + (size_t)b * 1024 * 1024;

    const int wpy   = wave >> 2;          // 0..1  : pixel row-group of 16
    const int wox   = wave & 3;           // 0..3  : o-group of 128
    const int ncol  = lane & 15;          // WMMA N / B-matrix column
    const int hi    = lane >> 4;          // lane half
    const int kb    = hi << 3;            // A-fragment K base (0 / 8)
    const int khalf = hi << 3;            // B-fragment K base (0 / 8)
    const int m     = (wpy << 4) + ncol;  // A-matrix row (pixel row in tile)

    v8f acc[8];
#pragma unroll
    for (int ob = 0; ob < 8; ++ob) {
        float bv = b1[wox * 128 + ob * 16 + ncol];
#pragma unroll
        for (int r = 0; r < 8; ++r) acc[ob][r] = bv;
    }

    // async transpose-stage of K-chunk kc into buffer buf
    auto stage = [&](int kc, int buf) {
        const float* gsrc = fb + (size_t)kc * 1024 + p0;
        for (int idx = tid; idx < 32 * KC; idx += 256) {
            int c = idx >> 5;            // channel within chunk
            int p = idx & 31;            // pixel (coalesced across a wave)
            unsigned la = (unsigned)(uintptr_t)&XtF[buf][p * XPF + c];
            async_cp_b32(la, gsrc + (size_t)c * 1024 + p);
        }
    };

    stage(0, 0);
    wait_async0();
    __syncthreads();

    for (int kcs = 0; kcs < 8; ++kcs) {          // 8 chunks of 128
        const int buf = kcs & 1;
        if (kcs < 7) stage((kcs + 1) * KC, buf ^ 1);   // overlap DMA with WMMA

        const float* xrow = &XtF[buf][m * XPF + kb];
        const __bf16* wbase = w1b + (size_t)(wox * 128 + ncol) * 1024 + kcs * KC + khalf;
#pragma unroll
        for (int kk = 0; kk < KC; kk += 32) {
            // A fragment: 16 fp32 from LDS -> bf16 (elements 0-7:K=kb+e, 8-15:K=kb+16+e)
            v4f f0 = *(const v4f*)(xrow + kk);
            v4f f1 = *(const v4f*)(xrow + kk + 4);
            v4f f2 = *(const v4f*)(xrow + kk + 16);
            v4f f3 = *(const v4f*)(xrow + kk + 20);
            v16bf a;
#pragma unroll
            for (int i = 0; i < 4; ++i) {
                a[i]      = (__bf16)f0[i];
                a[i + 4]  = (__bf16)f1[i];
                a[i + 8]  = (__bf16)f2[i];
                a[i + 12] = (__bf16)f3[i];
            }
#pragma unroll
            for (int ob = 0; ob < 8; ++ob) {
                const __bf16* wrow = wbase + (size_t)ob * 16 * 1024 + kk;
                v16bf bm = load_frag_b(wrow, wrow + 16);
                acc[ob] = __builtin_amdgcn_wmma_f32_16x16x32_bf16(
                    false, a, false, bm, (short)0, acc[ob], false, false);
            }
        }
        wait_async0();       // own async issues (next chunk) complete
        __syncthreads();     // everyone done computing buf before it is reused
    }

    // D layout: lane holds N=ncol, VGPR r holds M = r + 8*hi
    const size_t pg = (size_t)b * 1024 + p0;
#pragma unroll
    for (int ob = 0; ob < 8; ++ob) {
        int o = wox * 128 + ob * 16 + ncol;
#pragma unroll
        for (int r = 0; r < 8; ++r) {
            int prow = (wpy << 4) + (hi << 3) + r;
            float v = acc[ob][r];
            v = v > 0.f ? v : 0.f;
            h1[(pg + prow) * 512 + o] = (__bf16)v;
        }
    }
}

// ----------------------------- layer 2 -------------------------------------
// h2[p,o] = relu( h1[p,:] . w2[o,:] + b2[o] ), o=256. h1 row-major => A direct from global.
__global__ __launch_bounds__(256) void k_layer2(const __bf16* __restrict__ h1,
                                                const __bf16* __restrict__ w2b,
                                                const float* __restrict__ b2,
                                                __bf16* __restrict__ h2) {
    const int tid  = threadIdx.x;
    const int lane = tid & 31;
    const int wave = tid >> 5;
    const size_t p0 = (size_t)blockIdx.x * 32;   // 1024 WGs x 32 pixels

    const int wpy   = wave >> 2;
    const int wox   = wave & 3;                  // o-group of 64
    const int ncol  = lane & 15;
    const int hi    = lane >> 4;
    const int kb    = hi << 3;
    const int khalf = hi << 3;

    v8f acc[4];
#pragma unroll
    for (int ob = 0; ob < 4; ++ob) {
        float bv = b2[wox * 64 + ob * 16 + ncol];
#pragma unroll
        for (int r = 0; r < 8; ++r) acc[ob][r] = bv;
    }

    const __bf16* arow  = h1 + (p0 + (wpy << 4) + ncol) * 512 + kb;
    const __bf16* wbase = w2b + (size_t)(wox * 64 + ncol) * 512 + khalf;
#pragma unroll 2
    for (int kk = 0; kk < 512; kk += 32) {
        v16bf a = load_frag_b(arow + kk, arow + kk + 16);
#pragma unroll
        for (int ob = 0; ob < 4; ++ob) {
            const __bf16* wrow = wbase + (size_t)ob * 16 * 512 + kk;
            v16bf bm = load_frag_b(wrow, wrow + 16);
            acc[ob] = __builtin_amdgcn_wmma_f32_16x16x32_bf16(
                false, a, false, bm, (short)0, acc[ob], false, false);
        }
    }

#pragma unroll
    for (int ob = 0; ob < 4; ++ob) {
        int o = wox * 64 + ob * 16 + ncol;
#pragma unroll
        for (int r = 0; r < 8; ++r) {
            size_t prow = p0 + (wpy << 4) + (hi << 3) + r;
            float v = acc[ob][r];
            v = v > 0.f ? v : 0.f;
            h2[prow * 256 + o] = (__bf16)v;
        }
    }
}

// ----------------------------- layer 3 + cam blend --------------------------
// att[p] = 0.5 * ( sigmoid(h2[p,:].w3 + b3) + cam[p] )   (32->32 bilinear == identity)
__global__ __launch_bounds__(256) void k_att(const __bf16* __restrict__ h2,
                                             const __bf16* __restrict__ w3b,
                                             const float* __restrict__ b3,
                                             const float* __restrict__ cam,
                                             float* __restrict__ att) {
    int p = blockIdx.x * 256 + threadIdx.x;     // 32768 pixels
    const v8bf* hrow = (const v8bf*)(h2 + (size_t)p * 256);
    const v8bf* wrow = (const v8bf*)w3b;
    float s = 0.f;
#pragma unroll 4
    for (int ch = 0; ch < 32; ++ch) {
        v8bf hv = hrow[ch];
        v8bf wv = wrow[ch];
#pragma unroll
        for (int j = 0; j < 8; ++j) s += (float)hv[j] * (float)wv[j];
    }
    s += b3[0];
    float sig = 1.f / (1.f + __expf(-s));
    att[p] = 0.5f * (sig + cam[p]);
}

// ----------------------------- peak picking ---------------------------------
__global__ __launch_bounds__(128) void k_peaks(const float* __restrict__ att,
                                               int* __restrict__ peaks_i,    // [32][3][2] y1,x1
                                               float* __restrict__ coords) { // [32][3][4]
    __shared__ float s[1024];
    __shared__ float rv[128];
    __shared__ int   ri[128];
    __shared__ int   fyx[2];
    const int b = blockIdx.x, t = threadIdx.x;
    for (int i = t; i < 1024; i += 128) s[i] = att[b * 1024 + i];
    __syncthreads();

    for (int k = 0; k < 3; ++k) {
        float bestv = -1.f; int besti = 0;
        for (int i = t; i < 1024; i += 128) {
            float v = s[i];
            if (v > bestv) { bestv = v; besti = i; }
        }
        rv[t] = bestv; ri[t] = besti;
        __syncthreads();
        for (int off = 64; off > 0; off >>= 1) {
            if (t < off) {
                float v2 = rv[t + off]; int i2 = ri[t + off];
                if (v2 > rv[t] || (v2 == rv[t] && i2 < ri[t])) { rv[t] = v2; ri[t] = i2; }
            }
            __syncthreads();
        }
        if (t == 0) {
            int fi = ri[0]; float val = rv[0];
            int fy = fi >> 5, fx = fi & 31;
            if (!(val > 0.f)) { fy = 16; fx = 16; }
            fyx[0] = fy; fyx[1] = fx;
            int cy = fy * 16, cx = fx * 16;             // 512/32 scale, floor exact
            int y1 = min(max(cy - 192, 0), 128);        // clip to IMG-REGION = 128
            int x1 = min(max(cx - 192, 0), 128);
            peaks_i[(b * 3 + k) * 2 + 0] = y1;
            peaks_i[(b * 3 + k) * 2 + 1] = x1;
            float* cc = coords + (b * 3 + k) * 4;
            cc[0] = (float)x1; cc[1] = (float)y1;
            cc[2] = (float)(x1 + 384); cc[3] = (float)(y1 + 384);
        }
        __syncthreads();
        int fy = fyx[0], fx = fyx[1];
        for (int i = t; i < 1024; i += 128) {
            int y = i >> 5, x = i & 31;
            if (abs(y - fy) <= 5 && abs(x - fx) <= 5) s[i] = 0.f;
        }
        __syncthreads();
    }
}

// ----------------------------- region crop ----------------------------------
// x1 is always a multiple of 16 (fx*16-192 clipped to [0,128]) => rows are 16B
// aligned on both ends: copy with b128 (float4).
__global__ __launch_bounds__(192) void k_crop(const float* __restrict__ orig,
                                              const int* __restrict__ peaks_i,
                                              float* __restrict__ regions) {
    // 96 crops * 3 ch * 12 rowgroups = 3456 blocks; 192 thr = 2 rows x 96 float4
    const int blk = blockIdx.x;
    const int rg  = blk % 12;
    const int c   = (blk / 12) % 3;
    const int bk  = blk / 36;           // b*3 + k
    const int b   = bk / 3;
    const int y1  = peaks_i[bk * 2 + 0];
    const int x1  = peaks_i[bk * 2 + 1];
    const float* src = orig + ((size_t)b * 3 + c) * 512 * 512;
    float* dst = regions + ((size_t)bk * 3 + c) * 384 * 384;
    const int rsub = threadIdx.x / 96;   // 0/1: which row of the pair
    const int xq   = threadIdx.x % 96;   // float4 index within row
    for (int it = 0; it < 16; ++it) {
        int r = rg * 32 + it * 2 + rsub;
        const v4f* srow = (const v4f*)(src + (size_t)(y1 + r) * 512 + x1);
        v4f* drow = (v4f*)(dst + (size_t)r * 384);
        __builtin_prefetch(srow + 128, 0, 0);   // next row (512 floats ahead)
        drow[xq] = srow[xq];
    }
}

// ----------------------------- launcher --------------------------------------
extern "C" void kernel_launch(void* const* d_in, const int* in_sizes, int n_in,
                              void* d_out, int out_size, void* d_ws, size_t ws_size,
                              hipStream_t stream) {
    const float* features = (const float*)d_in[0];   // [32,1024,32,32]
    const float* cam      = (const float*)d_in[1];   // [32,32,32]
    const float* original = (const float*)d_in[2];   // [32,3,512,512]
    const float* w1 = (const float*)d_in[3];         // [512,1024]
    const float* b1 = (const float*)d_in[4];
    const float* w2 = (const float*)d_in[5];         // [256,512]
    const float* b2 = (const float*)d_in[6];
    const float* w3 = (const float*)d_in[7];         // [1,256]
    const float* b3 = (const float*)d_in[8];

    // workspace layout (bytes, 256-aligned)
    char* ws = (char*)d_ws;
    __bf16* w1b   = (__bf16*)(ws);                         // 1,048,576
    __bf16* w2b   = (__bf16*)(ws + 1048576);               //   262,144
    __bf16* w3b   = (__bf16*)(ws + 1310720);               //       512
    __bf16* h1    = (__bf16*)(ws + 1311232);               // 33,554,432
    __bf16* h2    = (__bf16*)(ws + 34865664);              // 16,777,216
    int*    peaks = (int*)   (ws + 51642880);              //       768

    // output layout: regions | coords | att
    float* out_regions = (float*)d_out;                            // 42,467,328
    float* out_coords  = out_regions + (size_t)96 * 3 * 384 * 384; //        384
    float* out_att     = out_coords + 384;                         //     32,768

    cvt_bf16<<<(524288 + 255) / 256, 256, 0, stream>>>(w1, w1b, 524288);
    cvt_bf16<<<(131072 + 255) / 256, 256, 0, stream>>>(w2, w2b, 131072);
    cvt_bf16<<<1, 256, 0, stream>>>(w3, w3b, 256);

    k_layer1<<<1024, 256, 0, stream>>>(features, w1b, b1, h1);
    k_layer2<<<1024, 256, 0, stream>>>(h1, w2b, b2, h2);
    k_att<<<128, 256, 0, stream>>>(h2, w3b, b3, cam, out_att);
    k_peaks<<<32, 128, 0, stream>>>(out_att, peaks, out_coords);
    k_crop<<<3456, 192, 0, stream>>>(original, peaks, out_regions);
}
